// ResNet_39831526703900
// MI455X (gfx1250) — matrix operand order repaired and checked by reading, not compiled, not run on hardware
//
#include <hip/hip_runtime.h>
#include <hip/hip_bf16.h>
#include <math.h>

// ---------------------------------------------------------------------------
// SPN (GateRecurrent2dnoind) pipeline for MI455X (gfx1250, wave32, WMMA, TDM).
//   conv3x3 (2->32)  [WMMA f16 GEMM, K=18 padded to 32]
//   spn_net pass 1   [4 directional scans, fused gate-normalize + fused max]
//   spn_net pass 2
//   conv3x3 (32->2) + log_softmax  [WMMA f16 GEMM, K=288 = 9 x 32]
// Horizontal scans stage 224x16 tiles into LDS via the Tensor Data Mover,
// using TDM's pad feature (pad 1 DWORD every 16) to produce a pitch-17
// bank-conflict-free layout directly.
// ---------------------------------------------------------------------------

#define BN   4
#define CG   384
#define CY   2
#define CHN  32
#define HH   224
#define WW   224
#define PLANE (HH * WW)          // 50176
#define NPIX  (BN * PLANE)       // 200704 pixels per channel-batch
#define NTILES (NPIX / 16)       // 12544 WMMA column tiles (exact)

typedef __attribute__((ext_vector_type(16))) _Float16 v16h;
typedef __attribute__((ext_vector_type(8)))  float    v8f;
typedef __attribute__((ext_vector_type(4))) unsigned int u32x4;
typedef __attribute__((ext_vector_type(8)))  int i32x8;
typedef __attribute__((ext_vector_type(4)))  int i32x4;

// ---------------------------------------------------------------------------
// Tensor Data Mover: async 2D tile load global->LDS.
// Tile: tile_dim0 = 16 DWORDs along W, tile_dim1 = 224 rows along H,
// row stride = WW. LDS pad: +1 DWORD after every 16 DWORDs -> pitch 17.
// Descriptor layout per CDNA5 ISA ch.8 (groups 0/1; groups 2/3 zero for 2D).
// This toolchain's builtin takes 6 args (g0, g1, g2, g3, extra, cpol).
// ---------------------------------------------------------------------------
__device__ __forceinline__ void tdm_load_tile16(const float* gsrc, void* ldst)
{
    const unsigned long long ga = (unsigned long long)(uintptr_t)gsrc;
    const unsigned lds_off = (unsigned)(unsigned long long)(uintptr_t)ldst;

    u32x4 g0;
    g0[0] = 1u;                                        // count=1, user mode
    g0[1] = lds_off;                                   // lds_addr (bytes)
    g0[2] = (unsigned)(ga & 0xFFFFFFFFull);            // global_addr[31:0]
    g0[3] = (unsigned)((ga >> 32) & 0x1FFFFFFull)      // global_addr[56:32]
          | (2u << 30);                                // type = 2 ("image")

    i32x8 g1;
    g1[0] = (int)((2u << 16)      // data_size = 4 bytes
                | (1u << 20)      // pad_enable
                | (3u << 22));    // pad_interval = 16 DWORDs (pad_amount=0 -> 1 DWORD)
    g1[1] = (int)(224u << 16);    // tensor_dim0 = 224 (low 16 in bits 63:48)
    g1[2] = (int)(224u << 16);    // tensor_dim0 hi = 0 | tensor_dim1 = 224 (low 16)
    g1[3] = (int)(16u  << 16);    // tensor_dim1 hi = 0 | tile_dim0 = 16
    g1[4] = (int)224u;            // tile_dim1 = 224 | tile_dim2 = 0
    g1[5] = (int)224u;            // tensor_dim0_stride[31:0] = 224
    g1[6] = 0;                    // stride hi | tensor_dim1_stride lo (unused, 2D)
    g1[7] = 0;

    const i32x4 z4 = {0, 0, 0, 0};
    const i32x8 z8 = {0, 0, 0, 0, 0, 0, 0, 0};
    __builtin_amdgcn_tensor_load_to_lds(g0, g1, z4, z4, z8, 0);
}

// ---------------------------------------------------------------------------
// Vertical scan: scan axis = H (t), lane axis = W (s). Coalesced per row.
// h(t) = (1-a1-a2-a3)*x + a1*h_prev[s-1] + a2*h_prev[s] + a3*h_prev[s+1]
// gates normalized on the fly by max(|a1|+|a2|+|a3|, 1).
// accum: out = max(out, h) (fused 4-direction combine).
// ---------------------------------------------------------------------------
__global__ __launch_bounds__(224) void scan_v(const float* __restrict__ x,
                                              const float* __restrict__ gates,
                                              float* __restrict__ out,
                                              int gi0, int reverse, int accum)
{
    const int s     = threadIdx.x;            // w coordinate 0..223
    const int plane = blockIdx.x;             // b*32 + c
    const int b = plane >> 5, c = plane & 31;

    const float* xp  = x   + (size_t)plane * PLANE;
    float*       op  = out + (size_t)plane * PLANE;
    const float* g1p = gates + ((size_t)b * CG + (size_t)(gi0 + 0) * CHN + c) * PLANE;
    const float* g2p = gates + ((size_t)b * CG + (size_t)(gi0 + 1) * CHN + c) * PLANE;
    const float* g3p = gates + ((size_t)b * CG + (size_t)(gi0 + 2) * CHN + c) * PLANE;

    __shared__ float hbuf[224];
    hbuf[s] = 0.0f;
    __syncthreads();

    for (int step = 0; step < HH; ++step) {
        const int t = reverse ? (HH - 1 - step) : step;
        const size_t off = (size_t)t * WW + s;

        float a1 = g1p[off], a2 = g2p[off], a3 = g3p[off];
        const float inv = 1.0f / fmaxf(fabsf(a1) + fabsf(a2) + fabsf(a3), 1.0f);
        a1 *= inv; a2 *= inv; a3 *= inv;
        const float xv = xp[off];

        const float hm = (s > 0)   ? hbuf[s - 1] : 0.0f;
        const float hc =             hbuf[s];
        const float hp = (s < 223) ? hbuf[s + 1] : 0.0f;
        __syncthreads();

        const float h = (1.0f - a1 - a2 - a3) * xv + a1 * hm + a2 * hc + a3 * hp;
        hbuf[s] = h;
        if (accum) op[off] = fmaxf(op[off], h);
        else       op[off] = h;
        __syncthreads();
    }
}

// ---------------------------------------------------------------------------
// Horizontal scan: scan axis = W (t), lane axis = H (s). Column access is
// stride-W in memory, so wave 0 issues four TDM tile loads (x,g1,g2,g3)
// into pitch-17 LDS tiles (conflict-free column reads; 17 coprime with 64
// banks), waits on TENSORcnt, and the workgroup barrier publishes the tile.
// The x tile is reused as the output tile and flushed coalesced.
// ---------------------------------------------------------------------------
#define HT 16
#define HP (HT + 1)

__global__ __launch_bounds__(224) void scan_h(const float* __restrict__ x,
                                              const float* __restrict__ gates,
                                              float* __restrict__ out,
                                              int gi0, int reverse, int accum)
{
    const int s     = threadIdx.x;            // h coordinate 0..223
    const int plane = blockIdx.x;
    const int b = plane >> 5, c = plane & 31;

    const float* xp  = x   + (size_t)plane * PLANE;
    float*       op  = out + (size_t)plane * PLANE;
    const float* g1p = gates + ((size_t)b * CG + (size_t)(gi0 + 0) * CHN + c) * PLANE;
    const float* g2p = gates + ((size_t)b * CG + (size_t)(gi0 + 1) * CHN + c) * PLANE;
    const float* g3p = gates + ((size_t)b * CG + (size_t)(gi0 + 2) * CHN + c) * PLANE;

    __shared__ float tx [224][HP];
    __shared__ float tg1[224][HP];
    __shared__ float tg2[224][HP];
    __shared__ float tg3[224][HP];
    __shared__ float hbuf[224];
    hbuf[s] = 0.0f;

    const int NT = WW / HT;                   // 14 tiles
    for (int tt = 0; tt < NT; ++tt) {
        const int tile = reverse ? (NT - 1 - tt) : tt;
        const int t0 = tile * HT;
        __syncthreads();                      // all LDS readers of prev tile done

        if (threadIdx.x < 32) {               // wave 0 issues the DMA (uniform)
            tdm_load_tile16(xp  + t0, &tx [0][0]);
            tdm_load_tile16(g1p + t0, &tg1[0][0]);
            tdm_load_tile16(g2p + t0, &tg2[0][0]);
            tdm_load_tile16(g3p + t0, &tg3[0][0]);
            __builtin_amdgcn_s_wait_tensorcnt(0);
        }
        __syncthreads();                      // tile visible to all waves

        for (int d = 0; d < HT; ++d) {
            const int dt = reverse ? (HT - 1 - d) : d;
            float a1 = tg1[s][dt], a2 = tg2[s][dt], a3 = tg3[s][dt];
            const float inv = 1.0f / fmaxf(fabsf(a1) + fabsf(a2) + fabsf(a3), 1.0f);
            a1 *= inv; a2 *= inv; a3 *= inv;
            const float xv = tx[s][dt];

            const float hm = (s > 0)   ? hbuf[s - 1] : 0.0f;
            const float hc =             hbuf[s];
            const float hp = (s < 223) ? hbuf[s + 1] : 0.0f;
            __syncthreads();

            const float h = (1.0f - a1 - a2 - a3) * xv + a1 * hm + a2 * hc + a3 * hp;
            hbuf[s] = h;
            tx[s][dt] = h;                    // reuse x tile as output tile
            __syncthreads();
        }

        for (int idx = s; idx < 224 * HT; idx += 224) {
            const int r  = idx / HT;
            const int cc = idx % HT;
            const size_t off = (size_t)r * WW + t0 + cc;
            if (accum) op[off] = fmaxf(op[off], tx[r][cc]);
            else       op[off] = tx[r][cc];
        }
    }
}

// ---------------------------------------------------------------------------
// conv_end_3: 3x3 conv 2->32, pad 1, via WMMA GEMM.
// D[m, n] = sum_k W[m,k] * im2col[k,n],  M=32 (2 tiles), K=18 (pad 32), N=NPIX.
// One wave per 16-pixel column tile; two WMMAs share the B fragment.
// ---------------------------------------------------------------------------
__global__ __launch_bounds__(256) void conv3_wmma(const float* __restrict__ y,
                                                  const float* __restrict__ w3,
                                                  float* __restrict__ out)
{
    const int lane  = threadIdx.x & 31;
    const int wave  = threadIdx.x >> 5;
    const int ntile = blockIdx.x * 8 + wave;      // grid = NTILES/8, exact
    const int nloc  = lane & 15;
    const int hi    = lane >> 4;                  // 0 or 1 (lane half)
    const int n     = ntile * 16 + nloc;
    const int bb    = n / PLANE;
    const int rem   = n % PLANE;
    const int oh    = rem / WW, ow = rem % WW;

    // B fragment: column n, K = hi*16 + j  (contiguous K order, same as A)
    v16h bfrag;
#pragma unroll
    for (int j = 0; j < 16; ++j) {
        const int K = hi * 16 + j;
        float v = 0.0f;
        if (K < 18) {
            const int ci = K / 9, r = K % 9;
            const int ih = oh + r / 3 - 1, iw = ow + r % 3 - 1;
            if (ih >= 0 && ih < HH && iw >= 0 && iw < WW)
                v = y[(((size_t)bb * CY + ci) * HH + ih) * WW + iw];
        }
        bfrag[j] = (_Float16)v;
    }

    // A fragments (weights), same contiguous K order as B:
    //   K = hi*16 + j,  row m = lane%16 (+16 for tile 1)
    v16h a0, a1;
#pragma unroll
    for (int j = 0; j < 16; ++j) {
        const int vv = j >> 1, tb = j & 1;
        const int K = hi * 16 + j;
        const int m = lane & 15;
        float w0 = 0.0f, w1 = 0.0f;
        if (K < 18) {
            w0 = w3[(size_t)(m     ) * 18 + K];
            w1 = w3[(size_t)(m + 16) * 18 + K];
        }
        a0[j] = (_Float16)w0;
        a1[j] = (_Float16)w1;
    }

    v8f c0 = {}, c1 = {};
    c0 = __builtin_amdgcn_wmma_f32_16x16x32_f16(false, a0, false, bfrag,
                                                (short)0, c0, false, false);
    c1 = __builtin_amdgcn_wmma_f32_16x16x32_f16(false, a1, false, bfrag,
                                                (short)0, c1, false, false);

    // D layout: VGPR r -> row m = r + hi*8, col = lane%16
#pragma unroll
    for (int r = 0; r < 8; ++r) {
        const int m0 = r + hi * 8;
        out[((size_t)bb * CHN + m0     ) * PLANE + rem] = c0[r];
        out[((size_t)bb * CHN + m0 + 16) * PLANE + rem] = c1[r];
    }
}

// ---------------------------------------------------------------------------
// conv_end_4: 3x3 conv 32->2 via WMMA (M=2 padded to 16, K=288 = 9x32),
// fused log_softmax over the 2 channels (lanes 0-15 hold both logits in
// acc[0]/acc[1] per the C/D layout).
// ---------------------------------------------------------------------------
__global__ __launch_bounds__(256) void conv4_wmma(const float* __restrict__ xin,
                                                  const float* __restrict__ w4,
                                                  float* __restrict__ out)
{
    const int lane  = threadIdx.x & 31;
    const int wave  = threadIdx.x >> 5;
    const int ntile = blockIdx.x * 8 + wave;
    const int nloc  = lane & 15;
    const int hi    = lane >> 4;
    const int n     = ntile * 16 + nloc;
    const int bb    = n / PLANE;
    const int rem   = n % PLANE;
    const int oh    = rem / WW, ow = rem % WW;

    v8f acc = {};
    for (int kc = 0; kc < 9; ++kc) {
        v16h bfrag, afrag;
#pragma unroll
        for (int j = 0; j < 16; ++j) {
            const int K = kc * 32 + hi * 16 + j;       // 0..287, always valid
            const int ci = K / 9, r = K % 9;
            const int ih = oh + r / 3 - 1, iw = ow + r % 3 - 1;
            float v = 0.0f;
            if (ih >= 0 && ih < HH && iw >= 0 && iw < WW)
                v = xin[(((size_t)bb * CHN + ci) * HH + ih) * WW + iw];
            bfrag[j] = (_Float16)v;
        }
#pragma unroll
        for (int j = 0; j < 16; ++j) {
            const int vv = j >> 1, tb = j & 1;
            const int K = kc * 32 + hi * 16 + j;
            const int m = lane & 15;
            const float w = (m < CY) ? w4[(size_t)m * 288 + K] : 0.0f;
            afrag[j] = (_Float16)w;
        }
        acc = __builtin_amdgcn_wmma_f32_16x16x32_f16(false, afrag, false, bfrag,
                                                     (short)0, acc, false, false);
    }

    if (hi == 0) {  // rows m=0 (acc[0]) and m=1 (acc[1]) live in lanes 0-15
        const float x0 = acc[0], x1 = acc[1];
        const float mx = fmaxf(x0, x1);
        const float lse = mx + logf(expf(x0 - mx) + expf(x1 - mx));
        out[((size_t)bb * CY + 0) * PLANE + rem] = x0 - lse;
        out[((size_t)bb * CY + 1) * PLANE + rem] = x1 - lse;
    }
}

// ---------------------------------------------------------------------------
// Driver. ws layout: buf0 = y32 / y2, buf1 = y1 (max-accumulated).
// Fused max-accumulate relies on stream-serialized kernel order.
// ---------------------------------------------------------------------------
extern "C" void kernel_launch(void* const* d_in, const int* in_sizes, int n_in,
                              void* d_out, int out_size, void* d_ws, size_t ws_size,
                              hipStream_t stream)
{
    const float* gates = (const float*)d_in[0];   // [4,384,224,224]
    const float* y     = (const float*)d_in[1];   // [4,2,224,224]
    const float* w3    = (const float*)d_in[2];   // [32,2,3,3]
    const float* w4    = (const float*)d_in[3];   // [2,32,3,3]
    float* out = (float*)d_out;                   // [4,2,224,224]

    const size_t TEN = (size_t)BN * CHN * PLANE;  // 6,422,528 floats
    float* buf0 = (float*)d_ws;                   // y32, then y2
    float* buf1 = buf0 + TEN;                     // y1

    const dim3 convGrid(NTILES / 8), convBlk(256);
    const dim3 scanGrid(BN * CHN), scanBlk(224);

    // conv_end_3: 2 -> 32
    conv3_wmma<<<convGrid, convBlk, 0, stream>>>(y, w3, buf0);

    // spn_net pass 1: lr (h,fwd) | rl (v,rev) | du (v,fwd) | ud (h,rev)
    scan_h<<<scanGrid, scanBlk, 0, stream>>>(buf0, gates, buf1, 0, 0, 0);
    scan_v<<<scanGrid, scanBlk, 0, stream>>>(buf0, gates, buf1, 3, 1, 1);
    scan_v<<<scanGrid, scanBlk, 0, stream>>>(buf0, gates, buf1, 6, 0, 1);
    scan_h<<<scanGrid, scanBlk, 0, stream>>>(buf0, gates, buf1, 9, 1, 1);

    // spn_net pass 2 (input y1 = buf1, output accumulates into buf0)
    scan_h<<<scanGrid, scanBlk, 0, stream>>>(buf1, gates, buf0, 0, 0, 0);
    scan_v<<<scanGrid, scanBlk, 0, stream>>>(buf1, gates, buf0, 3, 1, 1);
    scan_v<<<scanGrid, scanBlk, 0, stream>>>(buf1, gates, buf0, 6, 0, 1);
    scan_h<<<scanGrid, scanBlk, 0, stream>>>(buf1, gates, buf0, 9, 1, 1);

    // conv_end_4: 32 -> 2, fused log_softmax
    conv4_wmma<<<convGrid, convBlk, 0, stream>>>(buf0, w4, out);
}
